// ContextualAttention_89300960018511
// MI455X (gfx1250) — matrix operand order repaired and checked by reading, not compiled, not run on hardware
//
#include <hip/hip_runtime.h>
#include <hip/hip_bf16.h>

// ContextualAttention for MI455X (gfx1250, wave32, WMMA).
// Pipeline per sample (workspace reused across samples, stream-ordered):
//   k_prep : subsample b/f (stride 2) -> zero-padded channel-last bf16 planes [50][50][128]
//   k_norm : per-patch rnorm[i] = 1/sqrt(sum(w^2)+1152*EPS), and mask mm[i]
//   k_gemm1: yiT[j][i] = rnorm[i] * sum_{c,3x3} fs_patch[j] . bs_patch[i]   (bf16 WMMA, K=1152)
//   k_fsm  : double diagonal-band fuse (flat-index semantics incl. wraps) + masked softmax
//            over i, output yf[q][i] bf16
//   k_wt   : pre-gather conv-transpose weights wt[tap][c][i] = raw 4x4 patches of b (bf16)
//   k_conv : conv-transpose as GEMM (M=128 ch, N=pixels of one x-parity at fixed y,
//            K=2304 patches x 4 taps), bf16 WMMA, /4 epilogue -> out fp32

typedef unsigned short us16;
typedef __attribute__((ext_vector_type(16))) __bf16 v16bf;
typedef __attribute__((ext_vector_type(8)))  float  v8f;

#define L 2304      // 48*48 patch positions
#define HS 48
#define CCH 128
#define PW 50       // padded plane width (halo 1)

__device__ __forceinline__ us16 f32_to_bf16(float f) {
  union { float f; unsigned u; } x; x.f = f;
  unsigned r = x.u + 0x7FFFu + ((x.u >> 16) & 1u);   // RNE
  return (us16)(r >> 16);
}
__device__ __forceinline__ float bf16_to_f32(us16 h) {
  union { unsigned u; float f; } x; x.u = ((unsigned)h) << 16; return x.f;
}
__device__ __forceinline__ v8f v8f_zero() {
  v8f z = {0.f,0.f,0.f,0.f,0.f,0.f,0.f,0.f}; return z;
}

// ---------------- K1: pack subsampled, zero-padded, channel-last bf16 planes
__global__ void k_prep(const float* __restrict__ b, const float* __restrict__ f,
                       us16* __restrict__ bsp, us16* __restrict__ fsp) {
  int tid = blockIdx.x * 256 + threadIdx.x;
  if (tid >= PW * PW * CCH) return;
  int c = tid % CCH; int sp = tid / CCH;
  int wp = sp % PW;  int hp = sp / PW;
  float vb = 0.f, vf = 0.f;
  if (hp >= 1 && hp <= HS && wp >= 1 && wp <= HS) {
    int h = 2 * (hp - 1), w = 2 * (wp - 1);
    size_t off = (size_t)c * 9216 + h * 96 + w;
    vb = b[off]; vf = f[off];
  }
  bsp[tid] = f32_to_bf16(vb);
  fsp[tid] = f32_to_bf16(vf);
}

// ---------------- K2: patch reciprocal norms + interior mask (1 wave per patch)
__global__ void k_norm(const us16* __restrict__ bsp, const float* __restrict__ mask,
                       float* __restrict__ rnorm, float* __restrict__ mm) {
  int i = blockIdx.x; int lane = threadIdx.x;
  int h = i / HS, w = i % HS;
  float s = 0.f;
  for (int t = 0; t < 9; ++t) {
    int dh = t / 3, dw = t % 3;
    const us16* base = bsp + ((h + dh) * PW + (w + dw)) * CCH;
    for (int c = lane; c < CCH; c += 32) { float v = bf16_to_f32(base[c]); s += v * v; }
  }
  for (int m = 16; m; m >>= 1) s += __shfl_xor(s, m, 32);
  if (lane == 0) {
    rnorm[i] = 1.0f / sqrtf(s + 0.1152f);         // 1152 * EPS
    float msum = 0.f;
    for (int dh = -1; dh <= 1; ++dh)
      for (int dw = -1; dw <= 1; ++dw) {
        int hh = h + dh, ww = w + dw;
        if (hh >= 0 && hh < HS && ww >= 0 && ww < HS) msum += mask[(2 * hh) * 96 + 2 * ww];
      }
    mm[i] = ((msum / 9.0f) == 1.0f) ? 1.0f : 0.0f;
  }
}

// ---------------- K3: yiT[j][i] = rnorm[i] * <fs_patch[j], bs_patch[i]>
// WG tile 128j x 128i, 8 waves in 2(j)x4(i), K = 9 shifts x 4 channel-blocks of 32.
__global__ void k_gemm1(const us16* __restrict__ fsp, const us16* __restrict__ bsp,
                        const float* __restrict__ rnorm, float* __restrict__ yiT) {
  __shared__ us16 lA[128 * 32];   // [j-row][k] row = 64B
  __shared__ us16 lB[128 * 32];   // [i-row][k]
  const int tid = threadIdx.x, lane = tid & 31, wave = tid >> 5;
  const int waveJ = wave >> 2, waveI = wave & 3;
  const int itile = blockIdx.x, jtile = blockIdx.y;

  v8f acc[4][2];
  #pragma unroll
  for (int a = 0; a < 4; ++a)
    #pragma unroll
    for (int bb = 0; bb < 2; ++bb) acc[a][bb] = v8f_zero();

  // panel-load coordinates (2 16B chunks each for A and B per thread)
  int m0 = tid >> 2,            p0 = tid & 3;
  int m1 = (tid + 256) >> 2,    p1 = (tid + 256) & 3;
  int j0 = jtile * 128 + m0, hj0 = j0 / HS, wj0 = j0 % HS;
  int j1 = jtile * 128 + m1, hj1 = j1 / HS, wj1 = j1 % HS;
  int i0 = itile * 128 + m0, hi0 = i0 / HS, wi0 = i0 % HS;
  int i1 = itile * 128 + m1, hi1 = i1 / HS, wi1 = i1 % HS;

  for (int t9 = 0; t9 < 9; ++t9) {
    int dh = t9 / 3, dw = t9 % 3;
    for (int cb = 0; cb < 4; ++cb) {
      __syncthreads();
      *(uint4*)(lA + m0 * 32 + p0 * 8) =
        *(const uint4*)(fsp + ((hj0 + dh) * PW + (wj0 + dw)) * CCH + cb * 32 + p0 * 8);
      *(uint4*)(lA + m1 * 32 + p1 * 8) =
        *(const uint4*)(fsp + ((hj1 + dh) * PW + (wj1 + dw)) * CCH + cb * 32 + p1 * 8);
      *(uint4*)(lB + m0 * 32 + p0 * 8) =
        *(const uint4*)(bsp + ((hi0 + dh) * PW + (wi0 + dw)) * CCH + cb * 32 + p0 * 8);
      *(uint4*)(lB + m1 * 32 + p1 * 8) =
        *(const uint4*)(bsp + ((hi1 + dh) * PW + (wi1 + dw)) * CCH + cb * 32 + p1 * 8);
      __syncthreads();

      const int mA = lane & 15, s8 = (lane >> 4) * 8, koff = (lane >> 4) * 16;
      v16bf afr[4], bfr[2];
      #pragma unroll
      for (int jt = 0; jt < 4; ++jt) {            // A layout: K {0..7,16..23}/{8..15,24..31}
        int row = waveJ * 64 + jt * 16 + mA;
        union { v16bf v; uint4 q[2]; } u;
        u.q[0] = *(const uint4*)(lA + row * 32 + s8);
        u.q[1] = *(const uint4*)(lA + row * 32 + 16 + s8);
        afr[jt] = u.v;
      }
      #pragma unroll
      for (int it = 0; it < 2; ++it) {            // B layout: K {0..7,16..23}/{8..15,24..31}
        int row = waveI * 32 + it * 16 + mA;
        union { v16bf v; uint4 q[2]; } u;
        u.q[0] = *(const uint4*)(lB + row * 32 + s8);
        u.q[1] = *(const uint4*)(lB + row * 32 + 16 + s8);
        bfr[it] = u.v;
      }
      #pragma unroll
      for (int jt = 0; jt < 4; ++jt)
        #pragma unroll
        for (int it = 0; it < 2; ++it)
          acc[jt][it] = __builtin_amdgcn_wmma_f32_16x16x32_bf16(
              false, afr[jt], false, bfr[it], (short)0, acc[jt][it], false, false);
    }
  }
  // epilogue: C/D layout lane N = lane&15, M = v + 8*(lane>>4); scale column by rnorm[i]
  const int N = lane & 15, mhi = (lane >> 4) * 8;
  #pragma unroll
  for (int jt = 0; jt < 4; ++jt)
    #pragma unroll
    for (int it = 0; it < 2; ++it) {
      int icol = itile * 128 + waveI * 32 + it * 16 + N;
      float sc = rnorm[icol];
      #pragma unroll
      for (int v = 0; v < 8; ++v) {
        int jrow = jtile * 128 + waveJ * 64 + jt * 16 + v + mhi;
        yiT[(size_t)jrow * L + icol] = acc[jt][it][v] * sc;
      }
    }
}

// ---------------- K4: double band-fuse (exact flat-index semantics) + masked softmax
__global__ void k_fsm(const float* __restrict__ yiT, const float* __restrict__ mm,
                      us16* __restrict__ yf) {
  __shared__ float lg[L];
  __shared__ float red[256];
  const int q = blockIdx.x, tid = threadIdx.x;
  const int qh = q / HS, qw = q % HS, qT = qw * HS + qh;
  #pragma unroll
  for (int k = 0; k < 9; ++k) {
    int p = tid + k * 256;
    int ph = p / HS, pw = p % HS, pT = pw * HS + ph;
    float acc = 0.f;
    for (int d2 = -1; d2 <= 1; ++d2) {
      int aq = qT + d2, ap = pT + d2;
      if (aq < 0 || aq >= L || ap < 0 || ap >= L) continue;
      int b1 = (aq % HS) * HS + aq / HS;          // dec(): inverse transpose-flat
      int a1 = (ap % HS) * HS + ap / HS;
      for (int d1 = -1; d1 <= 1; ++d1) {
        int r = b1 + d1, cx = a1 + d1;
        if (r < 0 || r >= L || cx < 0 || cx >= L) continue;
        acc += yiT[(size_t)r * L + cx];           // mostly row-coalesced (L2-resident)
      }
    }
    lg[p] = acc * mm[p] * 10.0f;                  // SOFTMAX_SCALE
  }
  __syncthreads();
  float pm = -3.0e38f;
  #pragma unroll
  for (int k = 0; k < 9; ++k) pm = fmaxf(pm, lg[tid + k * 256]);
  red[tid] = pm; __syncthreads();
  for (int off = 128; off; off >>= 1) { if (tid < off) red[tid] = fmaxf(red[tid], red[tid + off]); __syncthreads(); }
  float mx = red[0]; __syncthreads();
  float ps = 0.f;
  #pragma unroll
  for (int k = 0; k < 9; ++k) { int p = tid + k * 256; float e = __expf(lg[p] - mx); lg[p] = e; ps += e; }
  red[tid] = ps; __syncthreads();
  for (int off = 128; off; off >>= 1) { if (tid < off) red[tid] += red[tid + off]; __syncthreads(); }
  float inv = 1.0f / red[0];
  #pragma unroll
  for (int k = 0; k < 9; ++k) {
    int p = tid + k * 256;
    yf[(size_t)q * L + p] = f32_to_bf16(lg[p] * inv * mm[p]);
  }
}

// ---------------- K5a: wt[tap][c][i] = raw_w[i][c][kh][kw] (bf16, coalesced writes)
__global__ void k_wt(const float* __restrict__ b, us16* __restrict__ wt) {
  int tid = blockIdx.x * 256 + threadIdx.x;
  if (tid >= 16 * CCH * L) return;
  int i = tid % L; int c = (tid / L) % CCH; int t = tid / (L * CCH);
  int kh = t >> 2, kw = t & 3;
  int hi = i / HS, wi = i % HS;
  int row = 2 * hi + kh - 1, col = 2 * wi + kw - 1;
  float v = 0.f;
  if (row >= 0 && row < 96 && col >= 0 && col < 96) v = b[(size_t)c * 9216 + row * 96 + col];
  wt[tid] = f32_to_bf16(v);
}

// ---------------- K5: conv-transpose as WMMA GEMM.
// Block = (y, x-parity): out[c, y, x=2n+px] for n=0..47, contraction over i x 4 taps.
__global__ void k_conv(const us16* __restrict__ wt, const us16* __restrict__ yf,
                       float* __restrict__ outp) {
  __shared__ us16 lA[128 * 32];   // weights [c][k]
  __shared__ us16 lB[48 * 32];    // yf      [n][k]
  const int y = blockIdx.x, px = blockIdx.y;
  const int tid = threadIdx.x, lane = tid & 31, wave = tid >> 5;
  v8f acc[3];
  #pragma unroll
  for (int nt = 0; nt < 3; ++nt) acc[nt] = v8f_zero();

  for (int uu = 0; uu < 2; ++uu) {
    int u = (y & 1) + 2 * uu;
    int h2 = y + u - 2;                       // even by construction
    if (h2 < 0) continue;
    int hrow = h2 >> 1;
    if (hrow >= HS) continue;
    for (int vv = 0; vv < 2; ++vv) {
      int v = px + 2 * vv;
      int dwv = px + vv - 1;                  // w = n + dwv
      int t_idx = (3 - u) * 4 + (3 - v);      // weight flip
      const us16* wbase = wt + (size_t)t_idx * CCH * L;
      for (int ib = 0; ib < 72; ++ib) {       // K chunks of 32 over i
        __syncthreads();
        #pragma unroll
        for (int r = 0; r < 2; ++r) {
          int e = tid + r * 256; int m = e >> 2, part = e & 3;
          *(uint4*)(lA + m * 32 + part * 8) =
            *(const uint4*)(wbase + (size_t)m * L + ib * 32 + part * 8);
        }
        if (tid < 192) {
          int n = tid >> 2, part = tid & 3;
          int w = n + dwv;
          uint4 vb = make_uint4(0u, 0u, 0u, 0u);
          if (w >= 0 && w < HS)
            vb = *(const uint4*)(yf + (size_t)(hrow * HS + w) * L + ib * 32 + part * 8);
          *(uint4*)(lB + n * 32 + part * 8) = vb;
        }
        __syncthreads();
        const int mA = lane & 15, s8 = (lane >> 4) * 8, koff = (lane >> 4) * 16;
        union { v16bf v; uint4 q[2]; } ua;
        int rowA = wave * 16 + mA;
        ua.q[0] = *(const uint4*)(lA + rowA * 32 + s8);
        ua.q[1] = *(const uint4*)(lA + rowA * 32 + 16 + s8);
        #pragma unroll
        for (int nt = 0; nt < 3; ++nt) {
          union { v16bf v; uint4 q[2]; } ub;
          int rn = nt * 16 + mA;
          ub.q[0] = *(const uint4*)(lB + rn * 32 + s8);
          ub.q[1] = *(const uint4*)(lB + rn * 32 + 16 + s8);
          acc[nt] = __builtin_amdgcn_wmma_f32_16x16x32_bf16(
              false, ua.v, false, ub.v, (short)0, acc[nt], false, false);
        }
      }
    }
  }
  const int N = lane & 15, mhi = (lane >> 4) * 8;
  #pragma unroll
  for (int nt = 0; nt < 3; ++nt) {
    int n = nt * 16 + N; int x = 2 * n + px;
    #pragma unroll
    for (int v8 = 0; v8 < 8; ++v8) {
      int c = wave * 16 + v8 + mhi;
      outp[((size_t)c * 96 + y) * 96 + x] = acc[nt][v8] * 0.25f;
    }
  }
}

extern "C" void kernel_launch(void* const* d_in, const int* in_sizes, int n_in,
                              void* d_out, int out_size, void* d_ws, size_t ws_size,
                              hipStream_t stream) {
  const float* b    = (const float*)d_in[0];
  const float* f    = (const float*)d_in[1];
  const float* mask = (const float*)d_in[2];
  float* out = (float*)d_out;
  char* ws = (char*)d_ws;

  // per-sample workspace (reused sequentially): ~42.6 MB
  const size_t OFF_BSP = 0;                       // 50*50*128 bf16 = 640,000 B
  const size_t OFF_FSP = 640000;                  // 640,000 B
  const size_t OFF_RN  = 1280000;                 // 2304 f32
  const size_t OFF_MM  = 1289216;                 // 2304 f32
  const size_t OFF_YI  = 1298432;                 // 2304*2304 f32 = 21,233,664 B
  const size_t OFF_YF  = 22532096;                // 2304*2304 bf16 = 10,616,832 B
  const size_t OFF_WT  = 33148928;                // 16*128*2304 bf16 = 9,437,184 B

  us16*  bsp   = (us16*)(ws + OFF_BSP);
  us16*  fsp   = (us16*)(ws + OFF_FSP);
  float* rnorm = (float*)(ws + OFF_RN);
  float* mm    = (float*)(ws + OFF_MM);
  float* yiT   = (float*)(ws + OFF_YI);
  us16*  yf    = (us16*)(ws + OFF_YF);
  us16*  wt    = (us16*)(ws + OFF_WT);

  for (int s = 0; s < 4; ++s) {
    const float* bs_ = b + (size_t)s * CCH * 96 * 96;
    const float* fs_ = f + (size_t)s * CCH * 96 * 96;
    const float* ms_ = mask + (size_t)s * 96 * 96;
    float* outs = out + (size_t)s * CCH * 96 * 96;

    k_prep<<<(PW * PW * CCH + 255) / 256, 256, 0, stream>>>(bs_, fs_, bsp, fsp);
    k_norm<<<L, 32, 0, stream>>>(bsp, ms_, rnorm, mm);
    k_gemm1<<<dim3(18, 18), 256, 0, stream>>>(fsp, bsp, rnorm, yiT);
    k_fsm<<<L, 256, 0, stream>>>(yiT, mm, yf);
    k_wt<<<(16 * CCH * L) / 256, 256, 0, stream>>>(bs_, wt);
    k_conv<<<dim3(96, 2), 256, 0, stream>>>(wt, yf, outs);
  }
}
